// CrossAttentionFusionMLP_83777632075953
// MI455X (gfx1250) — matrix operand order repaired and checked by reading, not compiled, not run on hardware
//
#include <hip/hip_runtime.h>
#include <math.h>

// ============================================================================
// CrossAttentionFusionMLP on MI455X (gfx1250), wave32 + WMMA.
//
// One wave32 = one sample. The 4 stylo tokens + 12 perp tokens are packed
// into a single 16-row x 32-col activation tile, so every dense linear in the
// model is a native v_wmma_f32_16x16x32_f16 (K=32=D, M=16=tokens), with f32
// accumulation. Attention softmax core (<=12 keys, head dim 8), LayerNorm,
// GELU and the classifier GEMV run on VALU via LDS.
// ============================================================================

typedef __attribute__((ext_vector_type(16))) _Float16 v16h;
typedef __attribute__((ext_vector_type(8)))  float    v8f;

#define DEV static __device__ __forceinline__

struct ParamTable { const float* p[104]; };

// ---------------------------------------------------------------------------
// Parameter index map. d_in[0]=stylo, d_in[1]=perp, then the params pytree
// flattened with alphabetical dict keys (JAX pytree order):
//   cls:  2 head_b, 3 head_w, 4 proj_b, 5 proj_w, 6 projln_b, 7 projln_g,
//         8..15 res1 {l1_b,l1_w,l2_b,l2_w,n1_b,n1_g,n2_b,n2_g}, 16..23 res2
//   layers: base 24 + 36*L, keys sorted: pc, ps, sc, ss
//     cross (pc@+0, sc@+18): ln_b, ln_g, bi, bo, wi, wo
//     slf   (ps@+6, ss@+24): ff1_b, ff1_w, ff2_b, ff2_w,
//                            ln_b, ln_g, ln2_b, ln2_g, bi, bo, wi, wo
//   perp_embed:  96 ln_b, 97 ln_g, 98 proj_b, 99 proj_w
//   stylo_embed: 100 ln_b, 101 ln_g, 102 proj_b, 103 proj_w
// ---------------------------------------------------------------------------

DEV float gelu_s(float x) {
    return 0.5f * x * (1.0f + erff(x * 0.7071067811865475f));
}

// ---- WMMA fragment builders (16x16x32 f16, per CDNA5 ISA 7.12.2) ----------

// A: 16x32 (MxK) f16 from LDS f32 (row stride ld). Lane&15 = row M;
// lanes 0-15 hold K = kc+[0..8)+[16..24), lanes 16-31 hold +8 shifted halves.
DEV v16h load_A(const float* lds, int ld, int kc, int lane) {
    const int row = lane & 15;
    const int kb  = kc + ((lane & 16) ? 8 : 0);
    const float4* q0 = reinterpret_cast<const float4*>(lds + row * ld + kb);
    const float4* q1 = reinterpret_cast<const float4*>(lds + row * ld + kb + 16);
    float4 a0 = q0[0], a1 = q0[1], a2 = q1[0], a3 = q1[1];
    v16h a;
    a[0]  = (_Float16)a0.x; a[1]  = (_Float16)a0.y;
    a[2]  = (_Float16)a0.z; a[3]  = (_Float16)a0.w;
    a[4]  = (_Float16)a1.x; a[5]  = (_Float16)a1.y;
    a[6]  = (_Float16)a1.z; a[7]  = (_Float16)a1.w;
    a[8]  = (_Float16)a2.x; a[9]  = (_Float16)a2.y;
    a[10] = (_Float16)a2.z; a[11] = (_Float16)a2.w;
    a[12] = (_Float16)a3.x; a[13] = (_Float16)a3.y;
    a[14] = (_Float16)a3.z; a[15] = (_Float16)a3.w;
    return a;
}

// B: 32x16 (KxN) tile = W^T with W row-major (Ntot x Ktot) in global memory.
// Lane&15 = output column n (within tile n0); lanes 0-15 hold K=kc+[0..8)+[16..24),
// lanes 16-31 hold +8 shifted halves (same order as load_A) -> 4x b128.
DEV v16h load_B(const float* __restrict__ W, int Ktot, int n0, int kc, int lane) {
    const int n  = n0 + (lane & 15);
    const int kb = kc + ((lane & 16) ? 8 : 0);
    const float4* q = reinterpret_cast<const float4*>(W + n * Ktot + kb);
    float4 b0 = q[0], b1 = q[1], b2 = q[4], b3 = q[5];
    v16h b;
    b[0]  = (_Float16)b0.x; b[1]  = (_Float16)b0.y;
    b[2]  = (_Float16)b0.z; b[3]  = (_Float16)b0.w;
    b[4]  = (_Float16)b1.x; b[5]  = (_Float16)b1.y;
    b[6]  = (_Float16)b1.z; b[7]  = (_Float16)b1.w;
    b[8]  = (_Float16)b2.x; b[9]  = (_Float16)b2.y;
    b[10] = (_Float16)b2.z; b[11] = (_Float16)b2.w;
    b[12] = (_Float16)b3.x; b[13] = (_Float16)b3.y;
    b[14] = (_Float16)b3.z; b[15] = (_Float16)b3.w;
    return b;
}

// C/D f32 16x16: lane&15 = column, rows r0..r0+8 (r0 = 8 for upper lane half).
DEV void store_C(float* lds, int ld, int n0, v8f c, const float* __restrict__ bias,
                 int lane) {
    const int col = n0 + (lane & 15);
    const int r0  = (lane & 16) ? 8 : 0;
    const float bb = bias[col];
#pragma unroll
    for (int i = 0; i < 8; ++i) lds[(r0 + i) * ld + col] = c[i] + bb;
}

// Y[16 x N] = X[16 x K] @ W^T + b  (K, N multiples of 32/16). All in LDS f32,
// weights f32 in global, math in f16 WMMA with f32 accumulation.
template<int K, int N>
DEV void linear_wmma(const float* X, int ldX, const float* __restrict__ W,
                     const float* __restrict__ b, float* Y, int ldY, int lane) {
#pragma unroll
    for (int t = 0; t < N / 16; ++t) {
        v8f c = {};
#pragma unroll
        for (int kc = 0; kc < K; kc += 32) {
            v16h a  = load_A(X, ldX, kc, lane);
            v16h bb = load_B(W, K, t * 16, kc, lane);
            c = __builtin_amdgcn_wmma_f32_16x16x32_f16(
                    false, a, false, bb, (short)0, c, false, false);
        }
        store_C(Y, ldY, t * 16, c, b, lane);
    }
    __builtin_amdgcn_wave_barrier();
}

// ---- attention core: P = [Q|K|V] 16x96 in LDS, heads=4, hd=8 --------------
template<int NQ, int NK>
DEV void attn_core(const float* P, int qlo, int klo, float* O, int lane) {
    const float scale = 0.3535533905932738f;  // 1/sqrt(8)
#pragma unroll
    for (int base = 0; base < NQ * 4; base += 32) {
        const int idx = base + lane;
        if (idx < NQ * 4) {
            const int q = qlo + (idx >> 2);
            const int h = idx & 3;
            const float* Qr = P + q * 96 + h * 8;
            float s[NK];
            float m = -1e30f;
#pragma unroll
            for (int k = 0; k < NK; ++k) {
                const float* Kr = P + (klo + k) * 96 + 32 + h * 8;
                float d = 0.f;
#pragma unroll
                for (int j = 0; j < 8; ++j) d += Qr[j] * Kr[j];
                d *= scale;
                s[k] = d;
                m = fmaxf(m, d);
            }
            float sum = 0.f;
#pragma unroll
            for (int k = 0; k < NK; ++k) { s[k] = __expf(s[k] - m); sum += s[k]; }
            const float inv = 1.f / sum;
            float o[8] = {0.f, 0.f, 0.f, 0.f, 0.f, 0.f, 0.f, 0.f};
#pragma unroll
            for (int k = 0; k < NK; ++k) {
                const float w = s[k] * inv;
                const float* Vr = P + (klo + k) * 96 + 64 + h * 8;
#pragma unroll
                for (int j = 0; j < 8; ++j) o[j] += w * Vr[j];
            }
            float* Or = O + q * 32 + h * 8;
#pragma unroll
            for (int j = 0; j < 8; ++j) Or[j] = o[j];
        }
    }
    __builtin_amdgcn_wave_barrier();
}

// Dst rows [rlo,rlo+nrows) = LayerNorm(X + R) over 32 features, one lane/row.
DEV void resln_rows(const float* X, const float* R, float* Dst,
                    int rlo, int nrows, const float* g, const float* b, int lane) {
    if (lane < nrows) {
        const int r = rlo + lane;
        float v[32];
        float m = 0.f;
#pragma unroll
        for (int f = 0; f < 32; ++f) { v[f] = X[r * 32 + f] + R[r * 32 + f]; m += v[f]; }
        m *= (1.f / 32.f);
        float var = 0.f;
#pragma unroll
        for (int f = 0; f < 32; ++f) { const float d = v[f] - m; var += d * d; }
        var *= (1.f / 32.f);
        const float inv = rsqrtf(var + 1e-5f);
#pragma unroll
        for (int f = 0; f < 32; ++f) Dst[r * 32 + f] = (v[f] - m) * inv * g[f] + b[f];
    }
    __builtin_amdgcn_wave_barrier();
}

// Full MHA (+ residual + LN) on packed 16x32 X; query rows [qlo,qlo+NQ),
// key/value rows [klo,klo+NK). Writes LN(X + attn) into Dst query rows.
template<int NQ, int NK>
DEV void mha_block(const float* X, float* Dst, int qlo, int klo,
                   const float* wi, const float* bi,
                   const float* wo, const float* bo,
                   const float* lng, const float* lnb,
                   float* region, int lane) {
    float* P = region;          // 16x96 = Q|K|V
    float* O = region + 1536;   // 16x32 attention output
    linear_wmma<32, 96>(X, 32, wi, bi, P, 96, lane);        // 6 WMMA
    attn_core<NQ, NK>(P, qlo, klo, O, lane);
    float* OP = region;         // reuse QKV space, 16x32
    linear_wmma<32, 32>(O, 32, wo, bo, OP, 32, lane);       // 2 WMMA
    resln_rows(X, OP, Dst, qlo, NQ, lng, lnb, lane);
}

// FeedForward (+ residual + LN2), in-place on X rows [rlo,rlo+nrows).
DEV void ff_block(float* X, int rlo, int nrows,
                  const float* w1, const float* b1,
                  const float* w2, const float* b2,
                  const float* lng, const float* lnb,
                  float* region, float* scratch, int lane) {
    float* H = region;          // 16x128 hidden
    linear_wmma<32, 128>(X, 32, w1, b1, H, 128, lane);      // 8 WMMA
    for (int i = lane; i < 16 * 128; i += 32) H[i] = gelu_s(H[i]);
    __builtin_amdgcn_wave_barrier();
    float* Y2 = scratch;        // 16x32
    linear_wmma<128, 32>(H, 128, w2, b2, Y2, 32, lane);     // 8 WMMA
    resln_rows(X, Y2, X, rlo, nrows, lng, lnb, lane);
}

// ---- classifier helpers (VALU GEMV; <10% of model FLOPs) ------------------
DEV void gemv(const float* x, int K, const float* __restrict__ W,
              const float* __restrict__ b, float* y, int lane) {
#pragma unroll
    for (int jj = 0; jj < 4; ++jj) {
        const int j = lane * 4 + jj;
        float a = b[j];
        for (int i = 0; i < K; ++i) a += x[i] * W[j * K + i];
        y[j] = a;
    }
    __builtin_amdgcn_wave_barrier();
}

DEV void ln128(float* x, const float* g, const float* b, int lane) {
    float m = 0.f, s2 = 0.f;
    for (int i = 0; i < 128; ++i) m += x[i];
    m *= (1.f / 128.f);
    for (int i = 0; i < 128; ++i) { const float d = x[i] - m; s2 += d * d; }
    s2 *= (1.f / 128.f);
    const float inv = rsqrtf(s2 + 1e-5f);
#pragma unroll
    for (int jj = 0; jj < 4; ++jj) {
        const int j = lane * 4 + jj;
        x[j] = (x[j] - m) * inv * g[j] + b[j];
    }
    __builtin_amdgcn_wave_barrier();
}

DEV void gelu128(float* x, int lane) {
#pragma unroll
    for (int jj = 0; jj < 4; ++jj) { const int j = lane * 4 + jj; x[j] = gelu_s(x[j]); }
    __builtin_amdgcn_wave_barrier();
}

// ===========================================================================
#define SLAB_FLOATS 3072   // X(512) + Xn(512) + region(2048)

__launch_bounds__(128, 1)
__global__ void caf_mlp_kernel(const float* __restrict__ stylo,
                               const float* __restrict__ perp,
                               float* __restrict__ out,
                               ParamTable PT) {
    __shared__ float smem[4 * SLAB_FLOATS];
    const int lane   = threadIdx.x & 31;
    const int wave   = threadIdx.x >> 5;
    const int sample = blockIdx.x * 4 + wave;
    const float* const* pp = PT.p;

    float* slab   = smem + wave * SLAB_FLOATS;
    float* bufX   = slab;          // 16x32
    float* bufY   = slab + 512;    // 16x32
    float* region = slab + 1024;   // 2048 floats, multi-purpose

    float* cur = bufX;
    float* alt = bufY;

    // ---- token embedding: Linear(1,32) + LayerNorm(32), one lane per token
    if (lane < 16) {
        const float *w, *b, *g, *lb;
        float val;
        if (lane < 4) {  // stylo tokens
            val = stylo[sample * 4 + lane];
            lb = pp[100]; g = pp[101]; b = pp[102]; w = pp[103];
        } else {         // perp tokens
            val = perp[sample * 12 + (lane - 4)];
            lb = pp[96]; g = pp[97]; b = pp[98]; w = pp[99];
        }
        float v[32];
        float m = 0.f;
#pragma unroll
        for (int f = 0; f < 32; ++f) { v[f] = val * w[f] + b[f]; m += v[f]; }
        m *= (1.f / 32.f);
        float var = 0.f;
#pragma unroll
        for (int f = 0; f < 32; ++f) { const float d = v[f] - m; var += d * d; }
        var *= (1.f / 32.f);
        const float inv = rsqrtf(var + 1e-5f);
#pragma unroll
        for (int f = 0; f < 32; ++f) cur[lane * 32 + f] = (v[f] - m) * inv * g[f] + lb[f];
    }
    __builtin_amdgcn_wave_barrier();

    // ---- 2 transformer layers -------------------------------------------
    for (int L = 0; L < 2; ++L) {
        const int lb = 24 + 36 * L;

        // sc: s_new = LN(s + MHA(q=s rows 0-3, kv=p rows 4-15)) -> alt rows 0-3
        {
            const int o = lb + 18;  // ln_b, ln_g, bi, bo, wi, wo
            mha_block<4, 12>(cur, alt, 0, 4,
                             pp[o + 4], pp[o + 2], pp[o + 5], pp[o + 3],
                             pp[o + 1], pp[o + 0], region, lane);
        }
        // pc: p_new = LN(p + MHA(q=p rows 4-15, kv=s rows 0-3)) -> alt rows 4-15
        {
            const int o = lb + 0;
            mha_block<12, 4>(cur, alt, 4, 0,
                             pp[o + 4], pp[o + 2], pp[o + 5], pp[o + 3],
                             pp[o + 1], pp[o + 0], region, lane);
        }
        { float* t = cur; cur = alt; alt = t; }

        // ss: self-attn + FF on stylo rows 0-3 (in place)
        {
            const int o = lb + 24;  // ff1_b,ff1_w,ff2_b,ff2_w,ln_b,ln_g,ln2_b,ln2_g,bi,bo,wi,wo
            mha_block<4, 4>(cur, cur, 0, 0,
                            pp[o + 10], pp[o + 8], pp[o + 11], pp[o + 9],
                            pp[o + 5], pp[o + 4], region, lane);
            ff_block(cur, 0, 4,
                     pp[o + 1], pp[o + 0], pp[o + 3], pp[o + 2],
                     pp[o + 7], pp[o + 6], region, alt, lane);
        }
        // ps: self-attn + FF on perp rows 4-15 (in place)
        {
            const int o = lb + 6;
            mha_block<12, 12>(cur, cur, 4, 4,
                              pp[o + 10], pp[o + 8], pp[o + 11], pp[o + 9],
                              pp[o + 5], pp[o + 4], region, lane);
            ff_block(cur, 4, 12,
                     pp[o + 1], pp[o + 0], pp[o + 3], pp[o + 2],
                     pp[o + 7], pp[o + 6], region, alt, lane);
        }
    }

    // ---- pooling: fused[0..32)=mean(s rows 0-3), [32..64)=mean(p rows 4-15)
    {
        float a = 0.f;
#pragma unroll
        for (int r = 0; r < 4; ++r) a += cur[r * 32 + lane];
        region[lane] = a * 0.25f;
        float c = 0.f;
#pragma unroll
        for (int r = 4; r < 16; ++r) c += cur[r * 32 + lane];
        region[32 + lane] = c * (1.f / 12.f);
    }
    __builtin_amdgcn_wave_barrier();

    // ---- classifier head -------------------------------------------------
    float* fused = region;        // 64
    float* xb    = region + 64;   // 128
    float* t1    = region + 192;  // 128
    float* t2    = region + 320;  // 128
    float* part  = region + 448;  // 32

    // proj 64->128 + LN + GELU
    gemv(fused, 64, pp[5], pp[4], xb, lane);
    ln128(xb, pp[7], pp[6], lane);
    gelu128(xb, lane);

    // two residual blocks: x = gelu(x + LN2(W2 gelu(LN1(W1 x))))
#pragma unroll
    for (int rb = 0; rb < 2; ++rb) {
        const int r = 8 + 8 * rb;  // l1_b,l1_w,l2_b,l2_w,n1_b,n1_g,n2_b,n2_g
        gemv(xb, 128, pp[r + 1], pp[r + 0], t1, lane);
        ln128(t1, pp[r + 5], pp[r + 4], lane);
        gelu128(t1, lane);
        gemv(t1, 128, pp[r + 3], pp[r + 2], t2, lane);
        ln128(t2, pp[r + 7], pp[r + 6], lane);
#pragma unroll
        for (int jj = 0; jj < 4; ++jj) {
            const int j = lane * 4 + jj;
            xb[j] = gelu_s(xb[j] + t2[j]);
        }
        __builtin_amdgcn_wave_barrier();
    }

    // head 128->1
    {
        const float* hw = pp[3];
        float a = 0.f;
#pragma unroll
        for (int jj = 0; jj < 4; ++jj) { const int j = lane * 4 + jj; a += xb[j] * hw[j]; }
        part[lane] = a;
        __builtin_amdgcn_wave_barrier();
        if (lane == 0) {
            float s = pp[2][0];
            for (int i = 0; i < 32; ++i) s += part[i];
            out[sample] = s;
        }
    }
}

// ===========================================================================
extern "C" void kernel_launch(void* const* d_in, const int* in_sizes, int n_in,
                              void* d_out, int out_size, void* d_ws, size_t ws_size,
                              hipStream_t stream) {
    (void)d_ws; (void)ws_size; (void)out_size;
    const float* stylo = (const float*)d_in[0];
    const float* perp  = (const float*)d_in[1];
    float* out = (float*)d_out;

    ParamTable PT;
    for (int i = 0; i < 104; ++i) PT.p[i] = (i < n_in) ? (const float*)d_in[i] : nullptr;

    const int B = in_sizes[0] / 4;          // 65536
    const int blocks = (B + 3) / 4;         // 4 samples (waves) per 128-thread block
    caf_mlp_kernel<<<blocks, 128, 0, stream>>>(stylo, perp, out, PT);
}
